// GraphAttentionLayer_66838281060959
// MI455X (gfx1250) — matrix operand order repaired and checked by reading, not compiled, not run on hardware
//
#include <hip/hip_runtime.h>
#include <hip/hip_bf16.h>
#include <math.h>

typedef __attribute__((ext_vector_type(16))) _Float16 v16h;
typedef __attribute__((ext_vector_type(8)))  float    v8f;

#define B_   8
#define N_   2048
#define F_   64
#define ALPHA 0.2f
#define NCHUNK (N_ / 32)   // 64 j-chunks of 32
#define NTILE  (N_ / 16)   // 128 i-tiles of 16

// ---------------------------------------------------------------------------
// Kernel 1: per (b, j-chunk of 32 rows):
//   Wh = h@W (fp32, in LDS), el = Wh@a1, er = Wh@a2,
//   and write Wh as f16 pre-swizzled into the WMMA B-matrix layout:
//   for chunk jc, col-block nb (16 cols), lane L (0..31), slot idx (0..15):
//     B[idx] = Wh[j0 + (idx/8)*16 + (L/16)*8 + idx%8][nb*16 + (L%16)]
// ---------------------------------------------------------------------------
__global__ __launch_bounds__(256)
void gat_wh_swizzle_kernel(const float* __restrict__ h,
                           const float* __restrict__ W,
                           const float* __restrict__ a,
                           float* __restrict__ el,
                           float* __restrict__ er,
                           _Float16* __restrict__ whsw) {
    const int b  = blockIdx.x / NCHUNK;
    const int jc = blockIdx.x % NCHUNK;
    const int j0 = jc * 32;
    const int t  = threadIdx.x;

    __shared__ float sW[F_ * F_];     // 16 KB
    __shared__ float sh[32 * F_];     // 8 KB
    __shared__ float swh[32 * F_];    // 8 KB

    for (int i = t; i < F_ * F_; i += 256) sW[i] = W[i];
    const float* hb = h + ((size_t)b * N_ + j0) * F_;
    for (int i = t; i < 32 * F_; i += 256) sh[i] = hb[i];
    __syncthreads();

    // 256 threads: each computes 8 outputs of one row (row = t/8, cols (t%8)*8..+8)
    {
        const int row = t >> 3;
        const int cb  = (t & 7) * 8;
        float acc[8];
#pragma unroll
        for (int f = 0; f < 8; ++f) acc[f] = 0.f;
        for (int k = 0; k < F_; ++k) {
            const float hv = sh[row * F_ + k];
#pragma unroll
            for (int f = 0; f < 8; ++f)
                acc[f] = fmaf(hv, sW[k * F_ + cb + f], acc[f]);
        }
#pragma unroll
        for (int f = 0; f < 8; ++f) swh[row * F_ + cb + f] = acc[f];
    }
    __syncthreads();

    // el / er for the 32 rows of this chunk
    if (t < 32) {
        float e1 = 0.f, e2 = 0.f;
#pragma unroll
        for (int f = 0; f < F_; ++f) {
            const float v = swh[t * F_ + f];
            e1 = fmaf(v, a[f], e1);
            e2 = fmaf(v, a[F_ + f], e2);
        }
        el[(size_t)b * N_ + j0 + t] = e1;
        er[(size_t)b * N_ + j0 + t] = e2;
    }

    // Swizzled f16 store: 2048 halfs per chunk (4 col-blocks x 32 lanes x 16 slots)
    _Float16* dst = whsw + (size_t)(b * NCHUNK + jc) * 2048;
#pragma unroll
    for (int u = 0; u < 8; ++u) {
        const int flat = t * 8 + u;
        const int nb   = flat >> 9;          // col-block
        const int rem  = flat & 511;
        const int L    = rem >> 4;           // lane
        const int idx  = rem & 15;           // slot
        const int kl   = ((idx >> 3) << 4) + ((L >> 4) << 3) + (idx & 7);        // K within chunk
        const int c    = (nb << 4) + (L & 15);         // column
        dst[flat] = (_Float16)swh[kl * F_ + c];
    }
}

// ---------------------------------------------------------------------------
// Kernel 1b: er_max[b] = max_j er[b, j]  (lrelu is monotonic -> row max of the
// score matrix is lrelu(el_i + er_max), so no online softmax is needed)
// ---------------------------------------------------------------------------
__global__ __launch_bounds__(256)
void gat_ermax_kernel(const float* __restrict__ er, float* __restrict__ ermax) {
    const int b = blockIdx.x;
    const int t = threadIdx.x;
    float m = -INFINITY;
    for (int i = t; i < N_; i += 256) m = fmaxf(m, er[(size_t)b * N_ + i]);
    __shared__ float red[256];
    red[t] = m;
    __syncthreads();
    for (int s = 128; s > 0; s >>= 1) {
        if (t < s) red[t] = fmaxf(red[t], red[t + s]);
        __syncthreads();
    }
    if (t == 0) ermax[b] = red[0];
}

// ---------------------------------------------------------------------------
// Kernel 2: fused softmax + (attention @ Wh) + ELU.
// One wave (32 threads) per (b, 16-row i-tile). Per j-chunk of 32:
//   - 4x float4 loads fetch the lane's two contiguous er runs
//     (A 16x32 layout: lane L -> row L%16; slots 0-7 -> K = (L/16)*8 + idx,
//      slots 8-15 -> K = 16 + (L/16)*8 + idx-8)
//   - p = exp(lrelu(el+er) - m) in f32, packed to the f16 A tile in registers
//   - 4 pre-swizzled B tiles loaded as one v16h per lane each (2x b128)
//   - 4x v_wmma_f32_16x16x32_f16 accumulate the 16x64 f32 output
// Then row-sum reduce (shfl), normalize, ELU, store.
// ---------------------------------------------------------------------------
__global__ __launch_bounds__(32)
void gat_attn_kernel(const float* __restrict__ el,
                     const float* __restrict__ er,
                     const float* __restrict__ ermax,
                     const _Float16* __restrict__ whsw,
                     float* __restrict__ out) {
    const int wid = blockIdx.x;          // 0 .. B*NTILE-1
    const int b   = wid >> 7;            // /128
    const int it  = wid & 127;
    const int i0  = it * 16;

    const int L = threadIdx.x;
    const int r = L & 15;                // A-matrix row for this lane
    const int h = L >> 4;                // lane half

    const float elr = el[(size_t)b * N_ + i0 + r];
    const float em  = ermax[b];
    const float s0  = elr + em;
    const float m   = fmaxf(s0, ALPHA * s0);   // row max (lrelu monotonic)

    const float* erb = er + (size_t)b * N_;
    const v16h*  wb  = (const v16h*)whsw + (size_t)b * NCHUNK * 4 * 32;

    v8f acc0 = {}, acc1 = {}, acc2 = {}, acc3 = {};
    float ls0 = 0.f, ls1 = 0.f, ls2 = 0.f, ls3 = 0.f;  // break the sum chain

#pragma unroll 2
    for (int jc = 0; jc < NCHUNK; ++jc) {
        // This lane's er values: two contiguous runs of 8 (k = h*8.. and 16+h*8..)
        const float4* er4 = (const float4*)(erb + jc * 32);
        const float4 e0 = er4[h * 2 + 0];   // k = h*8 + 0..3
        const float4 e1 = er4[h * 2 + 1];   // k = h*8 + 4..7
        const float4 e2 = er4[h * 2 + 4];   // k = 16 + h*8 + 0..3
        const float4 e3 = er4[h * 2 + 5];   // k = 16 + h*8 + 4..7
        float ev[16] = {e0.x, e0.y, e0.z, e0.w, e1.x, e1.y, e1.z, e1.w,
                        e2.x, e2.y, e2.z, e2.w, e3.x, e3.y, e3.z, e3.w};

        v16h A;
        float p[16];
#pragma unroll
        for (int idx = 0; idx < 16; ++idx) {
            float s = elr + ev[idx];
            s = fmaxf(s, ALPHA * s);         // leaky relu
            p[idx] = __expf(s - m);          // <= 1, no overflow
            A[idx] = (_Float16)p[idx];
        }
#pragma unroll
        for (int idx = 0; idx < 4; ++idx) {  // 4 independent partial sums
            ls0 += p[idx];
            ls1 += p[4 + idx];
            ls2 += p[8 + idx];
            ls3 += p[12 + idx];
        }

        const v16h* base = wb + (size_t)jc * 4 * 32 + L;
        const v16h B0 = base[0];
        const v16h B1 = base[32];
        const v16h B2 = base[64];
        const v16h B3 = base[96];
        acc0 = __builtin_amdgcn_wmma_f32_16x16x32_f16(false, A, false, B0, (short)0, acc0, false, false);
        acc1 = __builtin_amdgcn_wmma_f32_16x16x32_f16(false, A, false, B1, (short)0, acc1, false, false);
        acc2 = __builtin_amdgcn_wmma_f32_16x16x32_f16(false, A, false, B2, (short)0, acc2, false, false);
        acc3 = __builtin_amdgcn_wmma_f32_16x16x32_f16(false, A, false, B3, (short)0, acc3, false, false);
    }

    const float lsum = (ls0 + ls1) + (ls2 + ls3);
    // Full row sum: lanes r and r+16 hold complementary K halves of row r.
    const float total = lsum + __shfl_xor(lsum, 16, 32);
    // C/D layout: acc[q] in lane L is row M = q + 8*(L/16), col = (L%16)+nb*16.
    float rlM[8];
#pragma unroll
    for (int q = 0; q < 8; ++q) rlM[q] = 1.0f / __shfl(total, q + 8 * h, 32);

    float* ob = out + ((size_t)b * N_ + i0) * F_ + r;
#pragma unroll
    for (int q = 0; q < 8; ++q) {
        const int M = q + 8 * h;
        float v0 = acc0[q] * rlM[q];
        float v1 = acc1[q] * rlM[q];
        float v2 = acc2[q] * rlM[q];
        float v3 = acc3[q] * rlM[q];
        v0 = v0 > 0.f ? v0 : (__expf(v0) - 1.f);   // ELU
        v1 = v1 > 0.f ? v1 : (__expf(v1) - 1.f);
        v2 = v2 > 0.f ? v2 : (__expf(v2) - 1.f);
        v3 = v3 > 0.f ? v3 : (__expf(v3) - 1.f);
        float* row = ob + (size_t)M * F_;
        row[0]  = v0;
        row[16] = v1;
        row[32] = v2;
        row[48] = v3;
    }
}

// ---------------------------------------------------------------------------
extern "C" void kernel_launch(void* const* d_in, const int* in_sizes, int n_in,
                              void* d_out, int out_size, void* d_ws, size_t ws_size,
                              hipStream_t stream) {
    const float* h = (const float*)d_in[0];   // (8, 2048, 64)
    const float* W = (const float*)d_in[1];   // (64, 64)
    const float* a = (const float*)d_in[2];   // (128, 1)

    // Workspace layout (bytes):
    //   [0,       64K)  el   (B*N f32)
    //   [64K,    128K)  er   (B*N f32)
    //   [128K,   129K)  er_max (B f32, padded)
    //   [129K, 129K+2M) whsw (B*N*F f16, WMMA-B swizzled), 32B aligned
    float*    el    = (float*)d_ws;
    float*    er    = el + B_ * N_;
    float*    ermax = (float*)((char*)d_ws + 131072);
    _Float16* whsw  = (_Float16*)((char*)d_ws + 132096);

    gat_wh_swizzle_kernel<<<B_ * NCHUNK, 256, 0, stream>>>(h, W, a, el, er, whsw);
    gat_ermax_kernel<<<B_, 256, 0, stream>>>(er, ermax);
    gat_attn_kernel<<<B_ * NTILE, 32, 0, stream>>>(el, er, ermax, whsw, (float*)d_out);
}
